// Criterion_36464272343156
// MI455X (gfx1250) — compile-verified, matrix operand order checked by reading, not run on hardware
//
#include <hip/hip_runtime.h>
#include <hip/hip_bf16.h>

typedef __attribute__((ext_vector_type(16))) __bf16 v16bf;
typedef __attribute__((ext_vector_type(8)))  __bf16 v8bf;
typedef __attribute__((ext_vector_type(8)))  float  v8f;

static constexpr int BN   = 2048;   // batch
static constexpr int DN   = 8192;   // BCE feature dim
static constexpr int CN   = 1024;   // class dim (K of GEMM)
static constexpr float LOG_A = -7.6246189861593985f;   // -log(2048)

// ---------------- deterministic block reduction helpers ----------------
__device__ inline float block_sum256(float v, float* sm) {
    const int t = threadIdx.x;
    sm[t] = v; __syncthreads();
    #pragma unroll
    for (int s = 128; s > 0; s >>= 1) {
        if (t < s) sm[t] += sm[t + s];
        __syncthreads();
    }
    float r = sm[0]; __syncthreads();
    return r;
}
__device__ inline float block_max256(float v, float* sm) {
    const int t = threadIdx.x;
    sm[t] = v; __syncthreads();
    #pragma unroll
    for (int s = 128; s > 0; s >>= 1) {
        if (t < s) sm[t] = fmaxf(sm[t], sm[t + s]);
        __syncthreads();
    }
    float r = sm[0]; __syncthreads();
    return r;
}

// ---------------- init: zero scalars + f + g ----------------
__global__ __launch_bounds__(256) void init_kernel(float* scal, float* f, float* g) {
    const int idx = blockIdx.x * 256 + threadIdx.x;
    if (idx < 16) scal[idx] = 0.0f;
    if (idx < BN) { f[idx] = 0.0f; g[idx] = 0.0f; }
}

// ---------------- BCE partial sums (float4 streaming) ----------------
__global__ __launch_bounds__(256) void bce_part_kernel(const float4* __restrict__ x,
                                                       const float4* __restrict__ xt,
                                                       float* __restrict__ part, int n4) {
    __shared__ float sm[256];
    float acc = 0.0f;
    const int stride = gridDim.x * 256;
    for (int i = blockIdx.x * 256 + threadIdx.x; i < n4; i += stride) {
        const float4 a = x[i];
        const float4 b = xt[i];
        acc += a.x * logf(b.x) + (1.0f - a.x) * log1pf(-b.x);
        acc += a.y * logf(b.y) + (1.0f - a.y) * log1pf(-b.y);
        acc += a.z * logf(b.z) + (1.0f - a.z) * log1pf(-b.z);
        acc += a.w * logf(b.w) + (1.0f - a.w) * log1pf(-b.w);
    }
    const float tot = block_sum256(acc, sm);
    if (threadIdx.x == 0) part[blockIdx.x] = tot;
}

// ---------------- fp32 -> bf16 conversion ----------------
__global__ __launch_bounds__(256) void cvt_kernel(const float* __restrict__ src,
                                                  __bf16* __restrict__ dst, int n) {
    const int stride = gridDim.x * 256;
    for (int i = blockIdx.x * 256 + threadIdx.x; i < n; i += stride)
        dst[i] = (__bf16)src[i];
}

// ---------------- neg-entropy per target row ----------------
__global__ __launch_bounds__(256) void negent_kernel(const float* __restrict__ target,
                                                     float* __restrict__ negent) {
    __shared__ float sm[256];
    const int j = blockIdx.x;
    const float* row = target + (size_t)j * CN;
    float acc = 0.0f;
    #pragma unroll
    for (int k = 0; k < CN / 256; ++k) {
        const float t = row[threadIdx.x + k * 256];
        acc += (t > 0.0f) ? t * logf(t) : 0.0f;
    }
    const float tot = block_sum256(acc, sm);
    if (threadIdx.x == 0) negent[j] = tot;
}

// ---------------- WMMA bf16 GEMM: M = (negent[j] - logits·targetᵀ)/C, also Mᵀ ----------------
__device__ inline v16bf load_tile16x32(const __bf16* __restrict__ base, int row, int kcol, int kOff) {
    const __bf16* p = base + (size_t)row * CN + kcol + kOff;
    const v8bf lo = *(const v8bf*)p;          // K = kOff .. kOff+7
    const v8bf hi = *(const v8bf*)(p + 16);   // K = kOff+16 .. kOff+23
    v16bf r;
    #pragma unroll
    for (int e = 0; e < 8; ++e) { r[e] = lo[e]; r[e + 8] = hi[e]; }
    return r;
}

__global__ __launch_bounds__(256) void gemm_kernel(const __bf16* __restrict__ lb,
                                                   const __bf16* __restrict__ tb,
                                                   const float* __restrict__ negent,
                                                   float* __restrict__ M,
                                                   float* __restrict__ MT) {
    const int lane  = threadIdx.x & 31;
    const int wave  = threadIdx.x >> 5;
    const int waveI = wave & 3;          // 4 waves along i
    const int waveJ = wave >> 2;         // 2 waves along j
    const int iBase = blockIdx.y * 128 + waveI * 32;
    const int jBase = blockIdx.x * 64  + waveJ * 32;
    const int lrow  = lane & 15;
    const int kOff  = (lane < 16) ? 0 : 8;

    v8f acc[2][2];
    #pragma unroll
    for (int a = 0; a < 2; ++a)
        #pragma unroll
        for (int b = 0; b < 2; ++b)
            #pragma unroll
            for (int r = 0; r < 8; ++r) acc[a][b][r] = 0.0f;

    for (int k = 0; k < CN; k += 32) {
        v16bf A0 = load_tile16x32(lb, iBase + lrow,      k, kOff);
        v16bf A1 = load_tile16x32(lb, iBase + 16 + lrow, k, kOff);
        // B(K x N): B[k][n] = target[jBase+n][k]; row-major target == B's required per-lane layout
        v16bf B0 = load_tile16x32(tb, jBase + lrow,      k, kOff);
        v16bf B1 = load_tile16x32(tb, jBase + 16 + lrow, k, kOff);
        acc[0][0] = __builtin_amdgcn_wmma_f32_16x16x32_bf16(false, A0, false, B0, (short)0, acc[0][0], false, false);
        acc[0][1] = __builtin_amdgcn_wmma_f32_16x16x32_bf16(false, A0, false, B1, (short)0, acc[0][1], false, false);
        acc[1][0] = __builtin_amdgcn_wmma_f32_16x16x32_bf16(false, A1, false, B0, (short)0, acc[1][0], false, false);
        acc[1][1] = __builtin_amdgcn_wmma_f32_16x16x32_bf16(false, A1, false, B1, (short)0, acc[1][1], false, false);
    }

    // C/D layout: lanes 0-15 -> N=lane, VGPR r -> M=r ; lanes 16-31 -> N=lane-16, M=r+8
    const int n    = lane & 15;
    const int mOff = (lane < 16) ? 0 : 8;
    const float invC = 1.0f / (float)CN;
    #pragma unroll
    for (int ti = 0; ti < 2; ++ti) {
        #pragma unroll
        for (int tj = 0; tj < 2; ++tj) {
            const int j  = jBase + tj * 16 + n;
            const float ne = negent[j];
            #pragma unroll
            for (int r = 0; r < 8; ++r) {
                const int i = iBase + ti * 16 + mOff + r;
                const float m = (ne - acc[ti][tj][r]) * invC;
                M [(size_t)i * BN + j] = m;
                MT[(size_t)j * BN + i] = m;
            }
        }
    }
}

// ---------------- sum of M (partials) ----------------
__global__ __launch_bounds__(256) void sum_part_kernel(const float4* __restrict__ src,
                                                       float* __restrict__ part, int n4) {
    __shared__ float sm[256];
    float acc = 0.0f;
    const int stride = gridDim.x * 256;
    for (int i = blockIdx.x * 256 + threadIdx.x; i < n4; i += stride) {
        const float4 v = src[i];
        acc += v.x + v.y + v.z + v.w;
    }
    const float tot = block_sum256(acc, sm);
    if (threadIdx.x == 0) part[blockIdx.x] = tot;
}

// ---------------- deterministic 4096 -> scalar ----------------
__global__ __launch_bounds__(256) void reduce4096_kernel(const float* __restrict__ part,
                                                         float* __restrict__ dst) {
    __shared__ float sm[256];
    float acc = 0.0f;
    for (int k = threadIdx.x; k < 4096; k += 256) acc += part[k];
    const float tot = block_sum256(acc, sm);
    if (threadIdx.x == 0) *dst = tot;
}

__global__ void eps_kernel(float* scal) {
    scal[2] = 0.05f * scal[1] * (1.0f / (float)((size_t)BN * BN)) + 1e-8f;
}

// ---------------- Sinkhorn row/col LSE update (col uses MT) ----------------
// out[i] = eps*(log_a - LSE_j((in[j] - Msrc[i][j])/eps))
__global__ __launch_bounds__(256) void lse_row_kernel(const float* __restrict__ Msrc,
                                                      const float* __restrict__ vin,
                                                      float* __restrict__ vout,
                                                      const float* __restrict__ scal) {
    __shared__ float sm[256];
    const int i = blockIdx.x;
    const int t = threadIdx.x;
    const float eps = scal[2];
    const float inv_eps = 1.0f / eps;
    const float* row = Msrc + (size_t)i * BN;

    float v[8];
    float mx = -3.4e38f;
    #pragma unroll
    for (int k = 0; k < 8; ++k) {
        const int j = t + (k << 8);
        v[k] = (vin[j] - row[j]) * inv_eps;
        mx = fmaxf(mx, v[k]);
    }
    const float mAll = block_max256(mx, sm);
    float s = 0.0f;
    #pragma unroll
    for (int k = 0; k < 8; ++k) s += __expf(v[k] - mAll);
    const float sAll = block_sum256(s, sm);
    if (t == 0) vout[i] = eps * (LOG_A - (mAll + __logf(sAll)));
}

// ---------------- final <P, M> partials ----------------
__global__ __launch_bounds__(256) void final_part_kernel(const float4* __restrict__ M4,
                                                         const float* __restrict__ f,
                                                         const float* __restrict__ g,
                                                         const float* __restrict__ scal,
                                                         float* __restrict__ part) {
    __shared__ float sm[256];
    const float inv_eps = 1.0f / scal[2];
    float acc = 0.0f;
    const int n4 = (BN * BN) / 4;
    const int stride = gridDim.x * 256;
    for (int i4 = blockIdx.x * 256 + threadIdx.x; i4 < n4; i4 += stride) {
        const int base = i4 * 4;
        const int i = base >> 11;       // /2048
        const int j = base & (BN - 1);
        const float4 m = M4[i4];
        const float fi = f[i];
        acc += __expf((fi + g[j + 0] - m.x) * inv_eps) * m.x;
        acc += __expf((fi + g[j + 1] - m.y) * inv_eps) * m.y;
        acc += __expf((fi + g[j + 2] - m.z) * inv_eps) * m.z;
        acc += __expf((fi + g[j + 3] - m.w) * inv_eps) * m.w;
    }
    const float tot = block_sum256(acc, sm);
    if (threadIdx.x == 0) part[blockIdx.x] = tot;
}

__global__ void out_kernel(const float* __restrict__ scal, float* __restrict__ out) {
    const float bce = -scal[0] * (1.0f / (float)((size_t)BN * DN));
    out[0] = bce + scal[3];
}

// ---------------- launch ----------------
extern "C" void kernel_launch(void* const* d_in, const int* in_sizes, int n_in,
                              void* d_out, int out_size, void* d_ws, size_t ws_size,
                              hipStream_t stream) {
    const float* x      = (const float*)d_in[0];
    const float* xt     = (const float*)d_in[1];
    const float* logits = (const float*)d_in[2];
    const float* target = (const float*)d_in[3];
    float* out = (float*)d_out;

    char* ws = (char*)d_ws;
    float*  scal   = (float*)(ws + 0);                                   // 16 floats
    float*  negent = (float*)(ws + 256);                                 // 2048 f
    float*  f      = (float*)(ws + 8448);                                // 2048 f
    float*  g      = (float*)(ws + 16640);                               // 2048 f
    float*  part   = (float*)(ws + 24832);                               // 4096 f
    __bf16* lb     = (__bf16*)(ws + 65536);                              // 4 MiB
    __bf16* tb     = (__bf16*)(ws + 65536 + (size_t)4194304);            // 4 MiB
    float*  M      = (float*) (ws + 65536 + (size_t)8388608);            // 16 MiB
    float*  MT     = (float*) (ws + 65536 + (size_t)8388608 + (size_t)16777216); // 16 MiB

    init_kernel<<<8, 256, 0, stream>>>(scal, f, g);

    // BCE
    bce_part_kernel<<<4096, 256, 0, stream>>>((const float4*)x, (const float4*)xt,
                                              part, (BN * DN) / 4);
    reduce4096_kernel<<<1, 256, 0, stream>>>(part, &scal[0]);

    // cost matrix M via WMMA
    cvt_kernel<<<2048, 256, 0, stream>>>(logits, lb, BN * CN);
    cvt_kernel<<<2048, 256, 0, stream>>>(target, tb, BN * CN);
    negent_kernel<<<BN, 256, 0, stream>>>(target, negent);
    gemm_kernel<<<dim3(BN / 64, BN / 128), 256, 0, stream>>>(lb, tb, negent, M, MT);

    // eps = 0.05 * mean(M) + 1e-8
    sum_part_kernel<<<4096, 256, 0, stream>>>((const float4*)M, part, (BN * BN) / 4);
    reduce4096_kernel<<<1, 256, 0, stream>>>(part, &scal[1]);
    eps_kernel<<<1, 1, 0, stream>>>(scal);

    // 100 Sinkhorn iterations; column pass uses L2-resident MT so both passes are coalesced row-LSE
    for (int it = 0; it < 100; ++it) {
        lse_row_kernel<<<BN, 256, 0, stream>>>(M,  g, f, scal);
        lse_row_kernel<<<BN, 256, 0, stream>>>(MT, f, g, scal);
    }

    // sum(P * M)
    final_part_kernel<<<4096, 256, 0, stream>>>((const float4*)M, f, g, scal, part);
    reduce4096_kernel<<<1, 256, 0, stream>>>(part, &scal[3]);

    out_kernel<<<1, 1, 0, stream>>>(scal, out);
}
